// ProtoNet_3607772528648
// MI455X (gfx1250) — matrix operand order repaired and checked by reading, not compiled, not run on hardware
//
#include <hip/hip_runtime.h>
#include <hip/hip_bf16.h>

typedef __attribute__((ext_vector_type(16))) _Float16 v16h;
typedef __attribute__((ext_vector_type(8)))  float    v8f;

static constexpr int BATCH = 512;
static constexpr int LMAX  = 64;
static constexpr int DIM   = 128;
static constexpr int NPC   = 32;
static constexpr int PCS   = 16;
static constexpr int NFFN  = 256;

// Sinkhorn constants (EPS = 0.1), folded into base-2 domain for v_exp_f32/v_log_f32.
static constexpr float S_INV   = 14.4269504088896f;   // 1/(EPS*ln2)
static constexpr float EPS_LN2 = 0.0693147180560f;    // EPS*ln2
static constexpr float F_LOGA  = -0.27725887222397f;  // EPS * (-log 16)

__device__ __forceinline__ float fexp2(float x) {
#if __has_builtin(__builtin_amdgcn_exp2f)
  return __builtin_amdgcn_exp2f(x);      // v_exp_f32 (native exp2)
#else
  return exp2f(x);
#endif
}
__device__ __forceinline__ float flog2(float x) {
#if __has_builtin(__builtin_amdgcn_logf)
  return __builtin_amdgcn_logf(x);       // v_log_f32 (native log2)
#else
  return log2f(x);
#endif
}

// ---------------------------------------------------------------------------
// Prep: f32 -> f16 copies of atom_h and pc_X, plus row squared-norms.
// One wave per 128-element row; each lane handles 4 contiguous floats.
// ---------------------------------------------------------------------------
__global__ __launch_bounds__(256) void prep_kernel(
    const float* __restrict__ atom_h, const float* __restrict__ pc_X,
    _Float16* __restrict__ atomF16, _Float16* __restrict__ pcF16,
    float* __restrict__ an, float* __restrict__ pn) {
  const int NROWA = BATCH * LMAX;
  const int NROW  = NROWA + NPC * PCS;
  const int wave  = (int)((blockIdx.x * blockDim.x + threadIdx.x) >> 5);
  const int lane  = (int)(threadIdx.x & 31);
  if (wave >= NROW) return;

  const float* src; _Float16* dst; float* nrm;
  if (wave < NROWA) {
    src = atom_h + (size_t)wave * DIM;
    dst = atomF16 + (size_t)wave * DIM;
    nrm = an + wave;
  } else {
    const int r = wave - NROWA;
    src = pc_X + (size_t)r * DIM;
    dst = pcF16 + (size_t)r * DIM;
    nrm = pn + r;
  }

  const float4 v = ((const float4*)src)[lane];
  union { _Float16 h[4]; uint2 u; } pk;
  pk.h[0] = (_Float16)v.x; pk.h[1] = (_Float16)v.y;
  pk.h[2] = (_Float16)v.z; pk.h[3] = (_Float16)v.w;
  ((uint2*)dst)[lane] = pk.u;

  float s = v.x * v.x + v.y * v.y + v.z * v.z + v.w * v.w;
  s += __shfl_xor(s, 1, 32);
  s += __shfl_xor(s, 2, 32);
  s += __shfl_xor(s, 4, 32);
  s += __shfl_xor(s, 16, 32);
  s += __shfl_xor(s, 8, 32);
  if (lane == 0) *nrm = s;
}

// ---------------------------------------------------------------------------
// Sinkhorn: one 32-thread block (one wave) per (p, b) pair.
// Cost matrix C[16x64] built with v_wmma_f32_16x16x32_f16 and kept in VGPRs
// in the WMMA D layout: lane L owns rows m = 8*(L>>4)+r (r=0..7) and
// columns l = (L&15) + 16*t (t=0..3).
// ---------------------------------------------------------------------------
__global__ __launch_bounds__(32) void sinkhorn_kernel(
    const _Float16* __restrict__ atomF16, const _Float16* __restrict__ pcF16,
    const float* __restrict__ an, const float* __restrict__ pn,
    const int* __restrict__ n_atoms, float* __restrict__ mol) {
  const int p    = (int)blockIdx.x;
  const int b    = (int)blockIdx.y;
  const int lane = (int)threadIdx.x;
  const int c    = lane & 15;   // N index within 16-wide tile / M index for A rows
  const int h    = lane >> 4;   // lane-half selector

  union HV { uint4 q[2]; v16h v; };

  // --- GEMM: dots[m][l] = sum_d pc[p][m][d] * atom[b][l][d], K = 128 = 4x32 ---
  v8f acc[4] = {};
  const _Float16* Ar = pcF16 + (size_t)(p * PCS + c) * DIM;
  const _Float16* Bb = atomF16 + (size_t)b * LMAX * DIM;
#pragma unroll
  for (int kk = 0; kk < 4; ++kk) {
    // A 16x32 f16 layout (ISA 7.12.2): per lane-half, elems 0..7 -> K = 8h+0..7,
    // elems 8..15 -> K = 16+8h+0..7 (both contiguous 16B chunks).
    HV a;
    a.q[0] = *(const uint4*)(Ar + 32 * kk + 8 * h);
    a.q[1] = *(const uint4*)(Ar + 32 * kk + 16 + 8 * h);
#pragma unroll
    for (int t = 0; t < 4; ++t) {
      // B 32x16 f16 layout: lane n = c, elems 0..7 -> K = 8h+0..7, elems 8..15 -> K = 16+8h+0..7.
      const _Float16* Br = Bb + (size_t)(16 * t + c) * DIM + 32 * kk + 8 * h;
      HV bv;
      bv.q[0] = *(const uint4*)(Br);
      bv.q[1] = *(const uint4*)(Br + 16);
      acc[t] = __builtin_amdgcn_wmma_f32_16x16x32_f16(
          false, a.v, false, bv.v, (short)0, acc[t], false, false);
    }
  }

  const int   n    = n_atoms[b];
  const float logn = logf((float)n);

  float pnv[8];
  {
    const float4 p0 = *(const float4*)(pn + p * PCS + 8 * h);
    const float4 p1 = *(const float4*)(pn + p * PCS + 8 * h + 4);
    pnv[0] = p0.x; pnv[1] = p0.y; pnv[2] = p0.z; pnv[3] = p0.w;
    pnv[4] = p1.x; pnv[5] = p1.y; pnv[6] = p1.z; pnv[7] = p1.w;
  }

  float C[8][4], logb[4];
#pragma unroll
  for (int t = 0; t < 4; ++t) {
    const int  lt    = 16 * t + c;
    const bool valid = lt < n;
    const float a_l  = an[b * LMAX + lt];
    logb[t] = valid ? -logn : -1.0e9f;   // NEG for padded atoms
#pragma unroll
    for (int r = 0; r < 8; ++r) {
      const float ct = pnv[r] + a_l - 2.0f * acc[t][r];
      C[r][t] = valid ? ct : 1.0e6f;     // BIG for padded atoms
    }
  }

  float f[8], g[4];
#pragma unroll
  for (int r = 0; r < 8; ++r) f[r] = 0.0f;
#pragma unroll
  for (int t = 0; t < 4; ++t) g[t] = 0.0f;

  for (int it = 0; it < 30; ++it) {
    // f-update: row logsumexp over 64 columns (16 lanes of same half x 4 tiles)
#pragma unroll
    for (int r = 0; r < 8; ++r) {
      const float y0 = (g[0] - C[r][0]) * S_INV;
      const float y1 = (g[1] - C[r][1]) * S_INV;
      const float y2 = (g[2] - C[r][2]) * S_INV;
      const float y3 = (g[3] - C[r][3]) * S_INV;
      float mx = fmaxf(fmaxf(y0, y1), fmaxf(y2, y3));
      mx = fmaxf(mx, __shfl_xor(mx, 1, 32));
      mx = fmaxf(mx, __shfl_xor(mx, 2, 32));
      mx = fmaxf(mx, __shfl_xor(mx, 4, 32));
      mx = fmaxf(mx, __shfl_xor(mx, 8, 32));
      float sm = fexp2(y0 - mx) + fexp2(y1 - mx) + fexp2(y2 - mx) + fexp2(y3 - mx);
      sm += __shfl_xor(sm, 1, 32);
      sm += __shfl_xor(sm, 2, 32);
      sm += __shfl_xor(sm, 4, 32);
      sm += __shfl_xor(sm, 8, 32);
      f[r] = F_LOGA - EPS_LN2 * (mx + flog2(sm));
    }
    // g-update: column logsumexp over 16 rows (8 local + partner via xor 16)
#pragma unroll
    for (int t = 0; t < 4; ++t) {
      float y[8];
      float mx = -3.0e38f;
#pragma unroll
      for (int r = 0; r < 8; ++r) {
        y[r] = (f[r] - C[r][t]) * S_INV;
        mx = fmaxf(mx, y[r]);
      }
      mx = fmaxf(mx, __shfl_xor(mx, 16, 32));
      float sm = 0.0f;
#pragma unroll
      for (int r = 0; r < 8; ++r) sm += fexp2(y[r] - mx);
      sm += __shfl_xor(sm, 16, 32);
      g[t] = 0.1f * logb[t] - EPS_LN2 * (mx + flog2(sm));
    }
  }

  // dist = sum P*C with P = exp((f+g-C)/EPS)
  float dist = 0.0f;
#pragma unroll
  for (int t = 0; t < 4; ++t)
#pragma unroll
    for (int r = 0; r < 8; ++r)
      dist += fexp2((f[r] + g[t] - C[r][t]) * S_INV) * C[r][t];
  dist += __shfl_xor(dist, 1, 32);
  dist += __shfl_xor(dist, 2, 32);
  dist += __shfl_xor(dist, 4, 32);
  dist += __shfl_xor(dist, 8, 32);
  dist += __shfl_xor(dist, 16, 32);

  if (lane == 0)
    mol[b * NPC + p] = -dist * (float)PCS * (float)n * 0.01f;
}

// ---------------------------------------------------------------------------
// FFN: one wave per molecule. Lane handles 8 hidden units (256/32).
// ---------------------------------------------------------------------------
__global__ __launch_bounds__(256) void ffn_kernel(
    const float* __restrict__ mol, const float* __restrict__ W1,
    const float* __restrict__ b1, const float* __restrict__ W2,
    const float* __restrict__ b2, float* __restrict__ out) {
  const int wave = (int)((blockIdx.x * blockDim.x + threadIdx.x) >> 5);
  const int lane = (int)(threadIdx.x & 31);
  if (wave >= BATCH) return;

  const float* m = mol + wave * NPC;
  float acc = 0.0f;
#pragma unroll
  for (int j = 0; j < 8; ++j) {
    const int fidx = lane * 8 + j;
    float hh = b1[fidx];
#pragma unroll
    for (int p = 0; p < NPC; ++p) hh += m[p] * W1[p * NFFN + fidx];
    hh = fmaxf(hh, 0.0f);
    acc += hh * W2[fidx];
  }
  acc += __shfl_xor(acc, 1, 32);
  acc += __shfl_xor(acc, 2, 32);
  acc += __shfl_xor(acc, 4, 32);
  acc += __shfl_xor(acc, 8, 32);
  acc += __shfl_xor(acc, 16, 32);
  if (lane == 0) out[wave] = acc + b2[0];
}

// ---------------------------------------------------------------------------
extern "C" void kernel_launch(void* const* d_in, const int* in_sizes, int n_in,
                              void* d_out, int out_size, void* d_ws, size_t ws_size,
                              hipStream_t stream) {
  (void)in_sizes; (void)n_in; (void)out_size; (void)ws_size;
  const float* atom_h = (const float*)d_in[0];
  const float* pc_X   = (const float*)d_in[1];
  const float* W1     = (const float*)d_in[2];
  const float* b1     = (const float*)d_in[3];
  const float* W2     = (const float*)d_in[4];
  const float* b2     = (const float*)d_in[5];
  const int*   n_at   = (const int*)d_in[6];
  float* out = (float*)d_out;

  char* ws = (char*)d_ws;
  _Float16* atomF16 = (_Float16*)(ws);                                   // 8,388,608 B
  _Float16* pcF16   = (_Float16*)(ws + 8388608);                         //   131,072 B
  float*    an      = (float*)(ws + 8388608 + 131072);                   //   131,072 B
  float*    pn      = (float*)(ws + 8388608 + 2 * 131072);               //     2,048 B
  float*    mol     = (float*)(ws + 8388608 + 2 * 131072 + 2048);        //    65,536 B

  {
    const int waves  = BATCH * LMAX + NPC * PCS;   // 33,280 rows
    const int blocks = (waves + 7) / 8;            // 8 waves per 256-thread block
    prep_kernel<<<blocks, 256, 0, stream>>>(atom_h, pc_X, atomF16, pcF16, an, pn);
  }
  {
    dim3 grid(NPC, BATCH);
    sinkhorn_kernel<<<grid, 32, 0, stream>>>(atomF16, pcF16, an, pn, n_at, mol);
  }
  {
    ffn_kernel<<<BATCH / 8, 256, 0, stream>>>(mol, W1, b1, W2, b2, out);
  }
}
